// MambaSSM_85864986182280
// MI455X (gfx1250) — matrix operand order repaired and checked by reading, not compiled, not run on hardware
//
#include <hip/hip_runtime.h>
#include <hip/hip_bf16.h>
#include <math.h>

// ---------------------------------------------------------------------------
// Mamba SSM block for MI455X (gfx1250, wave32).
//  - All GEMMs via v_wmma_f32_16x16x32_bf16 (fp32 accumulate, fused epilogues)
//  - B-tiles staged into LDS by the Tensor Data Mover (double buffered),
//    gated by s_wait_tensorcnt; fragments read back with ds_load_b128.
//  - Register-resident recurrent scan (one channel per lane).
// ---------------------------------------------------------------------------

#define BATCH_L   1024      // B*L
#define D_MODEL_  1024
#define D_INNER_  2048
#define D_STATE_  16
#define D_CONV_   4
#define N_IN_     4096      // 2*D_INNER

typedef __attribute__((ext_vector_type(16))) __bf16  v16bf;
typedef __attribute__((ext_vector_type(8)))  float   v8f;
typedef __attribute__((ext_vector_type(4)))  unsigned int u32x4;
typedef __attribute__((ext_vector_type(8)))  int     i32x8;
typedef __attribute__((ext_vector_type(4)))  int     i32x4;

union FragBF {
    v16bf bf;
    u32x4 u[2];
};

#if defined(__gfx1250__) && __has_builtin(__builtin_amdgcn_tensor_load_to_lds)
#define USE_TDM 1
#else
#define USE_TDM 0
#endif

__device__ __forceinline__ unsigned short f32_to_bf16_rne(float f) {
    unsigned int u = __float_as_uint(f);
    unsigned int r = (u + 0x7FFFu + ((u >> 16) & 1u)) >> 16;
    return (unsigned short)r;
}

#if USE_TDM
// Issue one TDM 2D tile load (bf16 elements): tile (tileW x tileH) out of a
// row-major (nRows x rowLen) matrix starting at gsrc, into LDS at ldsAddr.
// Descriptor packing per CDNA5 ISA 8.3/8.4 (D# group0 + group1).
__device__ __forceinline__ void tdm_load_tile_bf16(
    const unsigned short* gsrc, unsigned ldsAddr,
    unsigned rowLen, unsigned nRows, unsigned tileW, unsigned tileH)
{
    unsigned long long ga = (unsigned long long)(const void*)gsrc;
    u32x4 g0;
    g0[0] = 1u;                                             // count=1, user D#
    g0[1] = ldsAddr;                                        // lds_addr
    g0[2] = (unsigned)ga;                                   // global_addr lo
    g0[3] = (unsigned)((ga >> 32) & 0x1FFFFFFu) | (2u << 30); // addr hi | type=2

    const unsigned d0 = (1u << 16);                         // data_size=2B
    const unsigned d1 = (rowLen & 0xFFFFu) << 16;           // tensor_dim0 lo16
    const unsigned d2 = ((rowLen >> 16) & 0xFFFFu) | ((nRows & 0xFFFFu) << 16);
    const unsigned d3 = ((nRows >> 16) & 0xFFFFu) | ((tileW & 0xFFFFu) << 16);
    const unsigned d4 = (tileH & 0xFFFFu);                  // tile_dim1 (tile_dim2=0)
    const unsigned d5 = rowLen;                             // dim0_stride lo32
    i32x8 g1 = { (int)d0, (int)d1, (int)d2, (int)d3, (int)d4, (int)d5, 0, 0 };
    i32x4 gz = { 0, 0, 0, 0 };
#if __clang_major__ >= 23
    i32x8 gz8 = { 0, 0, 0, 0, 0, 0, 0, 0 };
    __builtin_amdgcn_tensor_load_to_lds(g0, g1, gz, gz, gz8, 0);
#else
    __builtin_amdgcn_tensor_load_to_lds(g0, g1, gz, gz, 0);
#endif
}
#endif // USE_TDM

// ---------------------------------------------------------------------------
// fp32 -> bf16 bulk convert
// ---------------------------------------------------------------------------
__global__ void cvt_f32_to_bf16_kernel(const float* __restrict__ src,
                                       unsigned short* __restrict__ dst, int n) {
    int i = blockIdx.x * blockDim.x + threadIdx.x;
    if (i < n) dst[i] = f32_to_bf16_rne(src[i]);
}

// ---------------------------------------------------------------------------
// GEMM: C[M,N] = A[M,K] * Bw[N,K]^T   (A,Bw bf16 row-major, C fp32 row-major)
// MODE 0: plain store.  MODE 1: v = softplus(v + bias[col])   (for delta)
//
// Block = 8 waves stacked in M (128 rows); each wave owns a 16 x (NT*16)
// strip. The block-shared B tile (NT*16 rows x 32 K) is DMA'd into LDS by
// the TDM (double buffered), A fragments come straight from global (L2-hot).
// Grids are exact multiples -> no bounds checks anywhere in the hot loop.
// ---------------------------------------------------------------------------
template<int NT, int MODE>
__global__ __launch_bounds__(256) void gemm_bf16_wmma_kernel(
    const unsigned short* __restrict__ A,
    const unsigned short* __restrict__ Bw,
    float* __restrict__ C,
    int M, int N, int K,
    const float* __restrict__ bias)
{
    __shared__ __align__(16) unsigned short Btile[2][64 * 32];

    const int tid   = threadIdx.x;
    const int wave  = tid >> 5;          // 0..7
    const int lane  = tid & 31;
    const int half  = lane >> 4;         // 0/1
    const int r     = lane & 15;

    const int tileM  = blockIdx.y * 128 + wave * 16;
    const int tileN0 = blockIdx.x * (NT * 16);
    const int ksteps = K >> 5;

    const unsigned short* __restrict__ arow = A + (size_t)(tileM + r) * K;

    // ---- stage B tile for k-step 0 ----
#if USE_TDM
    if (wave == 0) {
        tdm_load_tile_bf16(Bw + (size_t)tileN0 * K,
                           (unsigned)(size_t)&Btile[0][0],
                           (unsigned)K, (unsigned)N, 32u, (unsigned)(NT * 16));
    }
#else
    for (int c = tid; c < NT * 64; c += 256) {
        const int row = c >> 2, ch = c & 3;
        *(u32x4*)&Btile[0][row * 32 + ch * 8] =
            *(const u32x4*)(Bw + (size_t)(tileN0 + row) * K + ch * 8);
    }
#endif

    v8f acc[NT] = {};

    for (int ks = 0; ks < ksteps; ++ks) {
        const int k0  = ks << 5;
        const int buf = ks & 1;

#if USE_TDM
        if (wave == 0) __builtin_amdgcn_s_wait_tensorcnt(0);
#endif
        __syncthreads();    // tile ks visible to all waves; prev buffer free

        // prefetch B tile for k-step ks+1 into the other buffer
        if (ks + 1 < ksteps) {
#if USE_TDM
            if (wave == 0) {
                tdm_load_tile_bf16(Bw + (size_t)tileN0 * K + (k0 + 32),
                                   (unsigned)(size_t)&Btile[buf ^ 1][0],
                                   (unsigned)K, (unsigned)N, 32u,
                                   (unsigned)(NT * 16));
            }
#else
            for (int c = tid; c < NT * 64; c += 256) {
                const int row = c >> 2, ch = c & 3;
                *(u32x4*)&Btile[buf ^ 1][row * 32 + ch * 8] =
                    *(const u32x4*)(Bw + (size_t)(tileN0 + row) * K +
                                    (k0 + 32) + ch * 8);
            }
#endif
        }

        // A fragment (16x32 bf16, ISA 7.12.2): half 0 -> K [0..7]+[16..23],
        // half 1 -> K [8..15]+[24..31] of row (tileM + r).
        FragBF a;
        a.u[0] = *(const u32x4*)(arow + k0 + 8 * half);
        a.u[1] = *(const u32x4*)(arow + k0 + 16 + 8 * half);
        if (ks + 1 < ksteps) __builtin_prefetch(arow + k0 + 32, 0, 0);

        // B fragments from LDS (ds_load_b128 x2 per tile): lane r = column
        // tileN0 + t*16 + r; half h holds K [8h..8h+7]+[16+8h..16+8h+7].
        FragBF b[NT];
#pragma unroll
        for (int t = 0; t < NT; ++t) {
            const unsigned short* bp = &Btile[buf][(t * 16 + r) * 32 + 8 * half];
            b[t].u[0] = *(const u32x4*)bp;
            b[t].u[1] = *(const u32x4*)(bp + 16);
        }

#pragma unroll
        for (int t = 0; t < NT; ++t) {
            acc[t] = __builtin_amdgcn_wmma_f32_16x16x32_bf16(
                false, a.bf, false, b[t].bf, (short)0, acc[t], false, false);
        }
    }

    // Epilogue + store.  acc[t][j] -> row = tileM + j + 8*half, col = tn + r.
#pragma unroll
    for (int t = 0; t < NT; ++t) {
        const int col = tileN0 + t * 16 + r;
#pragma unroll
        for (int j = 0; j < 8; ++j) {
            const int row = tileM + j + 8 * half;
            float v = acc[t][j];
            if (MODE == 1) {
                v += bias[col];
                v = (v > 20.0f) ? v : log1pf(__expf(v));   // softplus
            }
            C[(size_t)row * N + col] = v;
        }
    }
}

// ---------------------------------------------------------------------------
// Depthwise causal conv1d (k=4) + bias + SiLU.  Reads x_conv = xr[:, :2048].
// ---------------------------------------------------------------------------
__global__ void conv_silu_kernel(const float* __restrict__ xr,
                                 const float* __restrict__ cw,
                                 const float* __restrict__ cb,
                                 float* __restrict__ xc,
                                 unsigned short* __restrict__ xcbf)
{
    int idx = blockIdx.x * blockDim.x + threadIdx.x;
    if (idx >= BATCH_L * D_INNER_) return;
    const int l = idx >> 11;
    const int c = idx & (D_INNER_ - 1);

    float s = cb[c];
#pragma unroll
    for (int j = 0; j < D_CONV_; ++j) {
        const int li = l - (D_CONV_ - 1) + j;
        const float xv = (li >= 0) ? xr[(size_t)li * N_IN_ + c] : 0.0f;
        s += xv * cw[c * D_CONV_ + j];
    }
    const float sil = s / (1.0f + __expf(-s));
    xc[idx]   = sil;
    xcbf[idx] = f32_to_bf16_rne(sil);
}

// ---------------------------------------------------------------------------
// Recurrent scan: h[d,s] = exp(delta*A[s])*h + (delta*xc)*B[t,s];
//                 y[t,d] = sum_s C[t,s]*h[d,s].
// One channel per thread, h in 16 registers; B/C broadcast via LDS per step.
// ---------------------------------------------------------------------------
__global__ __launch_bounds__(256) void scan_kernel(
    const float* __restrict__ delta,
    const float* __restrict__ xc,
    const float* __restrict__ bc,     // [L, 32]  (:16 = B_proj, 16: = C_proj)
    const float* __restrict__ Avec,   // [16]
    float* __restrict__ y)
{
    __shared__ float sB[D_STATE_];
    __shared__ float sC[D_STATE_];

    const int d = blockIdx.x * 256 + threadIdx.x;

    float h[D_STATE_];
    float Ar[D_STATE_];
#pragma unroll
    for (int s = 0; s < D_STATE_; ++s) { h[s] = 0.0f; Ar[s] = Avec[s]; }

    for (int t = 0; t < BATCH_L; ++t) {
        if (threadIdx.x < 32) {
            const float v = bc[t * 32 + threadIdx.x];
            if (threadIdx.x < 16) sB[threadIdx.x]      = v;
            else                  sC[threadIdx.x - 16] = v;
        }
        __syncthreads();

        const float dt = delta[(size_t)t * D_INNER_ + d];
        const float dx = dt * xc[(size_t)t * D_INNER_ + d];
        float accv = 0.0f;
#pragma unroll
        for (int s = 0; s < D_STATE_; ++s) {
            const float dA = __expf(dt * Ar[s]);
            h[s] = dA * h[s] + dx * sB[s];
            accv += sC[s] * h[s];
        }
        y[(size_t)t * D_INNER_ + d] = accv;
        __syncthreads();
    }
}

// ---------------------------------------------------------------------------
// Gate: ytot = (y + xc*D) * silu(x_gate), emitted as bf16 for the last GEMM.
// ---------------------------------------------------------------------------
__global__ void gate_kernel(const float* __restrict__ y,
                            const float* __restrict__ xc,
                            const float* __restrict__ xr,
                            const float* __restrict__ Dvec,
                            unsigned short* __restrict__ ytotbf)
{
    int idx = blockIdx.x * blockDim.x + threadIdx.x;
    if (idx >= BATCH_L * D_INNER_) return;
    const int l = idx >> 11;
    const int c = idx & (D_INNER_ - 1);
    const float g   = xr[(size_t)l * N_IN_ + D_INNER_ + c];
    const float val = (y[idx] + xc[idx] * Dvec[c]) * (g / (1.0f + __expf(-g)));
    ytotbf[idx] = f32_to_bf16_rne(val);
}

// ---------------------------------------------------------------------------
// Launch
// ---------------------------------------------------------------------------
extern "C" void kernel_launch(void* const* d_in, const int* in_sizes, int n_in,
                              void* d_out, int out_size, void* d_ws, size_t ws_size,
                              hipStream_t stream) {
    const float* x      = (const float*)d_in[0];
    const float* W_in   = (const float*)d_in[1];
    const float* conv_w = (const float*)d_in[2];
    const float* conv_b = (const float*)d_in[3];
    const float* W_x    = (const float*)d_in[4];
    const float* W_dt   = (const float*)d_in[5];
    const float* b_dt   = (const float*)d_in[6];
    const float* W_out  = (const float*)d_in[7];
    const float* Avec   = (const float*)d_in[8];
    const float* Dvec   = (const float*)d_in[9];

    char*  ws  = (char*)d_ws;
    size_t off = 0;
    auto alloc = [&](size_t bytes) -> void* {
        void* p = ws + off;
        off = (off + bytes + 255) & ~(size_t)255;
        return p;
    };

    unsigned short* xBf    = (unsigned short*)alloc((size_t)BATCH_L * D_MODEL_ * 2);
    unsigned short* WinBf  = (unsigned short*)alloc((size_t)N_IN_   * D_MODEL_ * 2);
    unsigned short* WdtBf  = (unsigned short*)alloc((size_t)D_INNER_* D_INNER_ * 2);
    unsigned short* WxBf   = (unsigned short*)alloc((size_t)32      * D_INNER_ * 2);
    unsigned short* WoutBf = (unsigned short*)alloc((size_t)D_MODEL_* D_INNER_ * 2);
    float*          xr     = (float*)alloc((size_t)BATCH_L * N_IN_   * 4);
    float*          xc     = (float*)alloc((size_t)BATCH_L * D_INNER_* 4);
    unsigned short* xcBf   = (unsigned short*)alloc((size_t)BATCH_L * D_INNER_ * 2);
    float*          delta  = (float*)alloc((size_t)BATCH_L * D_INNER_* 4);
    float*          bc     = (float*)alloc((size_t)BATCH_L * 32 * 4);
    float*          ybuf   = (float*)alloc((size_t)BATCH_L * D_INNER_* 4);
    unsigned short* ytotBf = (unsigned short*)alloc((size_t)BATCH_L * D_INNER_ * 2);

    auto cvt = [&](const float* src, unsigned short* dst, int n) {
        cvt_f32_to_bf16_kernel<<<(n + 255) / 256, 256, 0, stream>>>(src, dst, n);
    };

    // 0) downconvert activations + weights to bf16
    cvt(x,     xBf,    BATCH_L * D_MODEL_);
    cvt(W_in,  WinBf,  N_IN_   * D_MODEL_);
    cvt(W_dt,  WdtBf,  D_INNER_* D_INNER_);
    cvt(W_x,   WxBf,   32      * D_INNER_);
    cvt(W_out, WoutBf, D_MODEL_* D_INNER_);

    // 1) xr = x @ W_in^T      (1024 x 4096, K=1024)
    gemm_bf16_wmma_kernel<4, 0><<<dim3(N_IN_ / 64, BATCH_L / 128), 256, 0, stream>>>(
        xBf, WinBf, xr, BATCH_L, N_IN_, D_MODEL_, nullptr);

    // 2) depthwise conv + SiLU
    conv_silu_kernel<<<(BATCH_L * D_INNER_ + 255) / 256, 256, 0, stream>>>(
        xr, conv_w, conv_b, xc, xcBf);

    // 3) delta = softplus(xc @ W_dt^T + b_dt)   (1024 x 2048, K=2048)
    gemm_bf16_wmma_kernel<4, 1><<<dim3(D_INNER_ / 64, BATCH_L / 128), 256, 0, stream>>>(
        xcBf, WdtBf, delta, BATCH_L, D_INNER_, D_INNER_, b_dt);

    // 4) bc = xc @ W_x^T                        (1024 x 32, K=2048)
    gemm_bf16_wmma_kernel<2, 0><<<dim3(1, BATCH_L / 128), 256, 0, stream>>>(
        xcBf, WxBf, bc, BATCH_L, 32, D_INNER_, nullptr);

    // 5) recurrent scan
    scan_kernel<<<D_INNER_ / 256, 256, 0, stream>>>(delta, xc, bc, Avec, ybuf);

    // 6) gating
    gate_kernel<<<(BATCH_L * D_INNER_ + 255) / 256, 256, 0, stream>>>(
        ybuf, xc, xr, Dvec, ytotBf);

    // 7) out = ytot @ W_out^T                   (1024 x 1024, K=2048)
    gemm_bf16_wmma_kernel<4, 0><<<dim3(D_MODEL_ / 64, BATCH_L / 128), 256, 0, stream>>>(
        ytotBf, WoutBf, (float*)d_out, BATCH_L, D_MODEL_, D_INNER_, nullptr);
}
